// GAT_34419867910941
// MI455X (gfx1250) — matrix operand order repaired and checked by reading, not compiled, not run on hardware
//
#include <hip/hip_runtime.h>
#include <hip/hip_bf16.h>
#include <math.h>

typedef __attribute__((ext_vector_type(16))) __bf16 v16bf;
typedef __attribute__((ext_vector_type(8)))  __bf16 v8bf;
typedef __attribute__((ext_vector_type(8)))  float  v8f;

#define NEG_SLOPE 0.2f
#define NGRP 4            // n-tiles per wave (register tiling in N)
#define WPB  8            // waves per block

__device__ __forceinline__ unsigned short f2bf_bits(float f) {
    unsigned u = __float_as_uint(f);
    u += 0x7FFFu + ((u >> 16) & 1u);     // round-to-nearest-even
    return (unsigned short)(u >> 16);
}

// ---------- f32 -> bf16 convert ----------
__global__ void k_cvt_bf16(const float* __restrict__ in, unsigned short* __restrict__ out, int n) {
    int i = blockIdx.x * blockDim.x + threadIdx.x;
    if (i < n) out[i] = f2bf_bits(in[i]);
}

// ---------- fill ----------
__global__ void k_fill(float* __restrict__ p, float v, int n) {
    int i = blockIdx.x * blockDim.x + threadIdx.x;
    if (i < n) p[i] = v;
}

// ---------- pack W [K,Nc] row-major f32 -> WMMA B-fragment bf16 layout ----------
// Wp index = ((nt*KT + kt)*32 + lane)*16 + j ; element = W[kt*32 + (lane>>4)*16 + j][nt*16 + (lane&15)]
__global__ void k_pack_w(const float* __restrict__ W, unsigned short* __restrict__ Wp,
                         int K, int Nc) {
    int tid = blockIdx.x * blockDim.x + threadIdx.x;
    int total = K * Nc;
    if (tid >= total) return;
    int KT = K >> 5;
    int j    = tid & 15;
    int lane = (tid >> 4) & 31;
    int t2   = tid >> 9;
    int kt   = t2 % KT;
    int nt   = t2 / KT;
    int ncol = nt * 16 + (lane & 15);
    int krow = kt * 32 + (lane >> 4) * 16 + j;
    Wp[tid] = f2bf_bits(W[(size_t)krow * Nc + ncol]);
}

// ---------- WMMA GEMM: C[M,Nc] = A[M,256]bf16 x Bpacked[256,Nc]bf16, f32 accumulate ----------
// Block = 8 waves (8 M-tiles), each wave computes 4 N-tiles (16x64 strip).
// The block's 32KB B-fragment slice is staged in LDS once and shared by all 8 waves.
__global__ void __launch_bounds__(WPB * 32)
k_gemm_wmma(const unsigned short* __restrict__ A,
            const unsigned short* __restrict__ Bp,
            float* __restrict__ C,
            int MT, int Nc) {
    const int K  = 256;
    const int KT = 8;                     // K / 32
    // LDS: NGRP n-tiles * KT k-steps * 32 lanes * 16 bf16 = 16384 bf16 = 32 KB
    __shared__ uint4 Bsh4[NGRP * KT * 32 * 16 / 8];
    unsigned short* Bsh = (unsigned short*)Bsh4;

    // cooperative copy of contiguous packed-B slice for this n-group
    {
        const uint4* gsrc = (const uint4*)(Bp + (size_t)blockIdx.y * (NGRP * KT * 32 * 16));
        const int nvec = NGRP * KT * 32 * 16 / 8;       // 2048 uint4
        for (int i = threadIdx.x; i < nvec; i += blockDim.x) Bsh4[i] = gsrc[i];
    }
    __syncthreads();

    int mTile = blockIdx.x * WPB + (threadIdx.x >> 5);
    int lane = threadIdx.x & 31;
    int l  = lane & 15;
    int hi = lane >> 4;
    if (mTile >= MT) return;

    // A fragment, contiguous K order matching packed B: lanes 0-15 rows, K chunk kb+hi*16..+15
    const unsigned short* arow = A + (size_t)(mTile * 16 + l) * K + hi * 16;

    v8f acc[NGRP] = {};
#pragma unroll
    for (int kt = 0; kt < KT; ++kt) {
        v8bf a0 = *(const v8bf*)(arow + kt * 32);
        v8bf a1 = *(const v8bf*)(arow + kt * 32 + 8);
        v16bf a = __builtin_shufflevector(a0, a1, 0,1,2,3,4,5,6,7,8,9,10,11,12,13,14,15);
        if (kt + 1 < KT) __builtin_prefetch(arow + (kt + 1) * 32, 0, 1);
#pragma unroll
        for (int nl = 0; nl < NGRP; ++nl) {
            const unsigned short* bf_ = Bsh + ((size_t)(nl * KT + kt) * 32 + lane) * 16;
            v8bf b0 = *(const v8bf*)(bf_);
            v8bf b1 = *(const v8bf*)(bf_ + 8);
            v16bf b = __builtin_shufflevector(b0, b1, 0,1,2,3,4,5,6,7,8,9,10,11,12,13,14,15);
            acc[nl] = __builtin_amdgcn_wmma_f32_16x16x32_bf16(
                false, a, false, b, (short)0, acc[nl], false, false);
        }
    }
    // D layout: lane (n=l, hi), VGPR r -> element (m = r + 8*hi, n)
#pragma unroll
    for (int nl = 0; nl < NGRP; ++nl) {
        float* crow = C + (size_t)(mTile * 16 + hi * 8) * Nc
                        + (size_t)(blockIdx.y * NGRP + nl) * 16 + l;
#pragma unroll
        for (int r = 0; r < 8; ++r) crow[(size_t)r * Nc] = acc[nl][r];
    }
}

// ---------- per-node attention logits: al_s[n,h] = <h[n,h,:], a_src[h,:]> ----------
__global__ void k_logits(const float* __restrict__ hfeat,
                         const float* __restrict__ asrc, const float* __restrict__ adst,
                         float* __restrict__ als, float* __restrict__ ald,
                         int N, int H, int C) {
    int tid = blockIdx.x * blockDim.x + threadIdx.x;
    if (tid >= N * H) return;
    int n = tid / H, hh = tid % H;
    const float* hp = hfeat + (size_t)n * H * C + (size_t)hh * C;
    const float* as = asrc + (size_t)hh * C;
    const float* ad = adst + (size_t)hh * C;
    float ss = 0.f, dd = 0.f;
    for (int c = 0; c < C; ++c) { float v = hp[c]; ss += v * as[c]; dd += v * ad[c]; }
    als[tid] = ss;
    ald[tid] = dd;
}

__device__ __forceinline__ void atomicMaxF(float* addr, float v) {
    if (v >= 0.f) atomicMax((int*)addr, __float_as_int(v));
    else          atomicMin((unsigned int*)addr, __float_as_uint(v));
}

// ---------- edge pass A: e = leakyrelu(al_s[src]+al_d[dst]); segment-max into m[dst] ----------
__global__ void k_edge_logit(const int* __restrict__ src, const int* __restrict__ dst,
                             int E, int Etot,
                             const float* __restrict__ als, const float* __restrict__ ald,
                             float* __restrict__ ebuf, float* __restrict__ mmax, int H) {
    int ei = blockIdx.x * blockDim.x + threadIdx.x;
    if (ei >= Etot) return;
    int s = (ei < E) ? src[ei] : (ei - E);
    int d = (ei < E) ? dst[ei] : (ei - E);
    for (int hh = 0; hh < H; ++hh) {
        float e = als[s * H + hh] + ald[d * H + hh];
        e = (e >= 0.f) ? e : NEG_SLOPE * e;
        ebuf[(size_t)ei * H + hh] = e;
        atomicMaxF(&mmax[d * H + hh], e);
    }
}

// ---------- edge pass B: ex = exp(e - m[dst]); segment-sum into s[dst] ----------
__global__ void k_edge_exp(const int* __restrict__ src, const int* __restrict__ dst,
                           int E, int Etot,
                           float* __restrict__ ebuf, const float* __restrict__ mmax,
                           float* __restrict__ ssum, int H) {
    int ei = blockIdx.x * blockDim.x + threadIdx.x;
    if (ei >= Etot) return;
    int d = (ei < E) ? dst[ei] : (ei - E);
    for (int hh = 0; hh < H; ++hh) {
        float ex = expf(ebuf[(size_t)ei * H + hh] - mmax[d * H + hh]);
        ebuf[(size_t)ei * H + hh] = ex;
        atomicAdd(&ssum[d * H + hh], ex);
    }
}

// ---------- edge pass C: agg[dst] += h[src] * alpha ; 4 features per thread ----------
__global__ void k_edge_msg(const int* __restrict__ src, const int* __restrict__ dst,
                           int E, int Etot,
                           const float* __restrict__ hfeat, const float* __restrict__ ebuf,
                           const float* __restrict__ ssum, float* __restrict__ agg,
                           int H, int C) {
    int tpe = (H * C) >> 2;                 // threads per edge (4 feats each)
    int tid = blockIdx.x * blockDim.x + threadIdx.x;
    if (tid >= Etot * tpe) return;
    int ei = tid / tpe;
    int part = tid % tpe;
    int cbase = part * 4;
    int hh = cbase / C;
    int s = (ei < E) ? src[ei] : (ei - E);
    int d = (ei < E) ? dst[ei] : (ei - E);
    float alpha = ebuf[(size_t)ei * H + hh] / (ssum[d * H + hh] + 1e-16f);
    const float4 hv = *(const float4*)(hfeat + (size_t)s * H * C + cbase);
    float* ap = agg + (size_t)d * H * C + cbase;
    atomicAdd(ap + 0, hv.x * alpha);
    atomicAdd(ap + 1, hv.y * alpha);
    atomicAdd(ap + 2, hv.z * alpha);
    atomicAdd(ap + 3, hv.w * alpha);
}

// ---------- epilogue (layers 1,2): bias + ELU + f32->bf16 for next GEMM ----------
__global__ void k_epi_elu_bf16(const float* __restrict__ agg, const float* __restrict__ b,
                               unsigned short* __restrict__ xbf, int total, int HC) {
    int i = blockIdx.x * blockDim.x + threadIdx.x;
    if (i >= total) return;
    float v = agg[i] + b[i % HC];
    v = (v > 0.f) ? v : expm1f(v);
    xbf[i] = f2bf_bits(v);
}

// ---------- epilogue (layer 3): bias, write f32 output ----------
__global__ void k_epi_out(const float* __restrict__ agg, const float* __restrict__ b,
                          float* __restrict__ out, int total, int C) {
    int i = blockIdx.x * blockDim.x + threadIdx.x;
    if (i >= total) return;
    out[i] = agg[i] + b[i % C];
}

static inline int cdiv(long a, long b) { return (int)((a + b - 1) / b); }

extern "C" void kernel_launch(void* const* d_in, const int* in_sizes, int n_in,
                              void* d_out, int out_size, void* d_ws, size_t ws_size,
                              hipStream_t stream) {
    const int N = 50000, E = 320000, FIN = 256, HID = 64, OUT = 64;
    const int Etot = E + N;
    const int MT = N / 16;   // 3125, exact

    const float* x   = (const float*)d_in[0];
    const int*   eix = (const int*)d_in[1];
    const int*   src = eix;
    const int*   dst = eix + E;
    const float* Wl[3]  = { (const float*)d_in[2], (const float*)d_in[6],  (const float*)d_in[10] };
    const float* asl[3] = { (const float*)d_in[3], (const float*)d_in[7],  (const float*)d_in[11] };
    const float* adl[3] = { (const float*)d_in[4], (const float*)d_in[8],  (const float*)d_in[12] };
    const float* bl[3]  = { (const float*)d_in[5], (const float*)d_in[9],  (const float*)d_in[13] };
    const int Hl[3] = { 4, 4, 1 };
    const int Cl[3] = { HID, HID, OUT };

    // workspace carve-out
    char* ws = (char*)d_ws;
    size_t off = 0;
    auto carve = [&](size_t bytes) { void* p = ws + off; off += (bytes + 255) & ~(size_t)255; return p; };
    unsigned short* XBF = (unsigned short*)carve((size_t)N * 256 * 2);   // bf16 layer input
    unsigned short* WP  = (unsigned short*)carve((size_t)256 * 256 * 2); // packed weights
    float* HB  = (float*)carve((size_t)N * 256 * 4);                     // h = X@W
    float* AGG = (float*)carve((size_t)N * 256 * 4);                     // aggregated messages
    float* ALS = (float*)carve((size_t)N * 4 * 4);
    float* ALD = (float*)carve((size_t)N * 4 * 4);
    float* MM  = (float*)carve((size_t)N * 4 * 4);
    float* SS  = (float*)carve((size_t)N * 4 * 4);
    float* EB  = (float*)carve((size_t)Etot * 4 * 4);
    (void)ws_size; (void)in_sizes; (void)n_in; (void)out_size;

    const int B = 256;

    // X -> bf16
    k_cvt_bf16<<<cdiv((long)N * FIN, B), B, 0, stream>>>(x, XBF, N * FIN);

    for (int layer = 0; layer < 3; ++layer) {
        const int H  = Hl[layer];
        const int C  = Cl[layer];
        const int Nc = H * C;          // 256, 256, 64
        const int NT = Nc / 16;

        // pack weights [256, Nc]
        k_pack_w<<<cdiv((long)256 * Nc, B), B, 0, stream>>>(Wl[layer], WP, 256, Nc);

        // init segment buffers
        k_fill<<<cdiv((long)N * Nc, B), B, 0, stream>>>(AGG, 0.f, N * Nc);
        k_fill<<<cdiv((long)N * H, B),  B, 0, stream>>>(SS, 0.f, N * H);
        k_fill<<<cdiv((long)N * H, B),  B, 0, stream>>>(MM, -INFINITY, N * H);

        // projection via WMMA: grid = (M-tile blocks, N-tile groups of 4)
        dim3 ggrid(cdiv(MT, WPB), NT / NGRP);
        k_gemm_wmma<<<ggrid, WPB * 32, 0, stream>>>(XBF, WP, HB, MT, Nc);

        // attention
        k_logits<<<cdiv((long)N * H, B), B, 0, stream>>>(HB, asl[layer], adl[layer], ALS, ALD, N, H, C);
        k_edge_logit<<<cdiv(Etot, B), B, 0, stream>>>(src, dst, E, Etot, ALS, ALD, EB, MM, H);
        k_edge_exp<<<cdiv(Etot, B), B, 0, stream>>>(src, dst, E, Etot, EB, MM, SS, H);
        k_edge_msg<<<cdiv((long)Etot * (H * C / 4), B), B, 0, stream>>>(
            src, dst, E, Etot, HB, EB, SS, AGG, H, C);

        // epilogue
        if (layer < 2) {
            k_epi_elu_bf16<<<cdiv((long)N * Nc, B), B, 0, stream>>>(AGG, bl[layer], XBF, N * Nc, Nc);
        } else {
            k_epi_out<<<cdiv((long)N * OUT, B), B, 0, stream>>>(AGG, bl[layer], (float*)d_out, N * OUT, OUT);
        }
    }
}
